// MVM_Model_76510547411496
// MI455X (gfx1250) — compile-verified, matrix operand order checked by reading, not run on hardware
//
#include <hip/hip_runtime.h>
#include <stdint.h>

// ---------------------------------------------------------------------------
// CDNA5 (gfx1250) wave32 WMMA types
// ---------------------------------------------------------------------------
typedef __attribute__((ext_vector_type(16))) _Float16 v16h;
typedef __attribute__((ext_vector_type(8)))  float    v8f;

#define DEV __device__ __forceinline__

DEV _Float16 u2h(uint32_t u) { union { unsigned short s; _Float16 h; } x; x.s = (unsigned short)u; return x.h; }
DEV unsigned short h2u(_Float16 h) { union { unsigned short s; _Float16 h; } x; x.h = h; return x.s; }

// quant(x, 16, 13): symmetric fixed point, 13 fractional bits
DEV float quant13(float x) {
    float t = rintf(x * 8192.0f);
    t = fminf(fmaxf(t, -32768.0f), 32767.0f);
    return t * (1.0f / 8192.0f);
}
// quant(x, 32, 24): accumulator fixed point, 24 fractional bits
DEV float quant24(float x) {
    float t = rintf(x * 16777216.0f);
    t = fminf(fmaxf(t, -2147483648.0f), 2147483647.0f);
    return t * (1.0f / 16777216.0f);
}
// Exact two-fp16 split of a (quantized) f32 value: x == hi + lo
DEV uint32_t pack_split(float x) {
    _Float16 h = (_Float16)x;
    _Float16 l = (_Float16)(x - (float)h);
    return (uint32_t)h2u(h) | ((uint32_t)h2u(l) << 16);
}
DEV float unpack_pair(uint32_t p) {
    return (float)u2h(p & 0xffffu) + (float)u2h(p >> 16);
}

enum { F_BN = 1, F_ADD = 2, F_RELU = 4, F_W32 = 8, F_WPAIR = 16, F_FC = 32 };

// ---------------------------------------------------------------------------
// Implicit-GEMM conv via v_wmma_f32_16x16x32_f16 (split-fp16, 3 WMMAs / term)
// One wave (32 threads) computes a 64(M=Cout) x 16(N=pixel) output block:
// 4 M-tiles share one gathered B fragment -> 12 WMMAs per activation gather.
// Compile-time KH/KW turns the k->(ci,kh,kw) decode into mul-shift constants.
//
// A (weights): lane holds row m=lane&15, k = lb+{0..7} and 16+lb+{0..7},
//              lb = (lane>>4)*8   (ISA 16-bit A 16x32 layout)
// B (inputs) : lane holds col n=lane&15, same k striping (mirror layout)
// C/D (f32)  : vgpr r -> (m = r + 8*(lane>>4), n = lane&15)
// ---------------------------------------------------------------------------
template <int KH, int KW>
__global__ __launch_bounds__(32)
void conv_wmma_kernel(const uint32_t* __restrict__ act,   // split-pair input, NCHW
                      const uint32_t* __restrict__ wgt,   // split-pair weights [CoutPad][Kpad]
                      const float*    __restrict__ bnS,
                      const float*    __restrict__ bnB,   // also FC bias
                      const float*    __restrict__ res,
                      float*          __restrict__ out32,
                      uint32_t*       __restrict__ outPair,
                      int Cin, int IH, int IW, int Cout, int OH, int OW,
                      int stride, int pad,
                      int Kdim, int Kpad, int nTiles, int flags)
{
    constexpr int KHW = KH * KW;
    const int tile = blockIdx.x;
    const int mT4 = tile / nTiles, nT = tile % nTiles;   // mT4: 64-row block
    const int lane = threadIdx.x;
    const int lb  = (lane >> 4) << 3;
    const int l15 = lane & 15;
    const int mRowBase = (mT4 << 6) + l15;
    const int nCol = (nT << 4) + l15;

    const int hw = OH * OW;
    const int b  = nCol / hw;
    const int rp = nCol - b * hw;
    const int oh = rp / OW;
    const int ow = rp - oh * OW;
    const int ihB = oh * stride - pad;
    const int iwB = ow * stride - pad;
    const uint32_t* wrow0 = wgt + (size_t)mRowBase * (size_t)Kpad;
    const size_t rowStride16 = (size_t)16 * (size_t)Kpad;
    const size_t planeB = (size_t)b * (size_t)Cin;

    v8f acc0 = {}, acc1 = {}, acc2 = {}, acc3 = {};
    for (int kb = 0; kb < Kpad; kb += 32) {
        // ---- gather one B fragment (shared by 4 M-tiles) ----
        uint32_t bw[16];
#pragma unroll
        for (int e = 0; e < 16; ++e) {
            const int ko = lb + (e < 8 ? e : e + 8);
            const int k  = kb + ko;
            uint32_t bv = 0u;
            if (k < Kdim) {                        // zero-fill K padding
                const int ci = k / KHW;            // compile-time constant divisor
                const int r2 = k - ci * KHW;
                const int kh = r2 / KW;
                const int kw = r2 - kh * KW;
                const int ih = ihB + kh;
                const int iw = iwB + kw;
                if ((unsigned)ih < (unsigned)IH && (unsigned)iw < (unsigned)IW)
                    bv = act[((planeB + ci) * IH + ih) * IW + iw];
            }
            bw[e] = bv;
        }
        v16h bh, bl;
#pragma unroll
        for (int e = 0; e < 16; ++e) {
            bh[e] = u2h(bw[e] & 0xffffu);
            bl[e] = u2h(bw[e] >> 16);
        }
        // ---- 4 M-tiles: load A pair, 3 split-product WMMAs each ----
#pragma unroll
        for (int j = 0; j < 4; ++j) {
            const uint32_t* wr = wrow0 + (size_t)j * rowStride16 + kb;
            uint32_t aw[16];
#pragma unroll
            for (int e = 0; e < 16; ++e)
                aw[e] = wr[lb + (e < 8 ? e : e + 8)];   // contiguous 8-dword runs -> b128
            v16h ah, al;
#pragma unroll
            for (int e = 0; e < 16; ++e) {
                ah[e] = u2h(aw[e] & 0xffffu);
                al[e] = u2h(aw[e] >> 16);
            }
            v8f a = (j == 0) ? acc0 : (j == 1) ? acc1 : (j == 2) ? acc2 : acc3;
            a = __builtin_amdgcn_wmma_f32_16x16x32_f16(false, ah, false, bh, (short)0, a, false, false);
            a = __builtin_amdgcn_wmma_f32_16x16x32_f16(false, ah, false, bl, (short)0, a, false, false);
            a = __builtin_amdgcn_wmma_f32_16x16x32_f16(false, al, false, bh, (short)0, a, false, false);
            if (j == 0) acc0 = a; else if (j == 1) acc1 = a; else if (j == 2) acc2 = a; else acc3 = a;
        }
    }

    // ---- epilogue ----
#pragma unroll
    for (int j = 0; j < 4; ++j) {
        const v8f a = (j == 0) ? acc0 : (j == 1) ? acc1 : (j == 2) ? acc2 : acc3;
        const int mBase = (mT4 << 6) + (j << 4) + ((lane >> 4) << 3);
#pragma unroll
        for (int r8 = 0; r8 < 8; ++r8) {
            const int m = mBase + r8;
            float v = a[r8];
            if (flags & F_FC) {                    // quant(mm + bias, 32, 24) -> d_out[b][m]
                if (m < Cout) {
                    v = quant24(v + bnB[m]);
                    out32[(size_t)nCol * (size_t)Cout + m] = v;
                }
                continue;
            }
            v = quant24(v);                        // crossbar accumulator quant
            if (flags & F_BN) v = v * bnS[m] + bnB[m];
            const size_t oidx = (((size_t)b * Cout + m) * OH + oh) * OW + ow;
            if (flags & F_ADD)  v += res[oidx];
            if (flags & F_RELU) v = fmaxf(v, 0.0f);
            if (flags & F_W32)  out32[oidx] = v;
            if (flags & F_WPAIR) outPair[oidx] = pack_split(quant13(v));
        }
    }
}

// ---------------------------------------------------------------------------
// Weight repack: f32 OIHW -> quant(16,13) split pairs, zero-padded [CoutPad][Kpad]
// ---------------------------------------------------------------------------
__global__ void repack_kernel(const float* __restrict__ w, uint32_t* __restrict__ dst,
                              int Cout, int K, int Kpad, long long n)
{
    long long i = (long long)blockIdx.x * blockDim.x + threadIdx.x;
    if (i >= n) return;
    int m = (int)(i / Kpad);
    int k = (int)(i - (long long)m * Kpad);
    float v = (m < Cout && k < K) ? w[(size_t)m * (size_t)K + k] : 0.0f;
    dst[i] = pack_split(quant13(v));
}

// (x - mean)/std -> quant(16,13) split pairs
__global__ void preprocess_kernel(const float* __restrict__ x, uint32_t* __restrict__ out, long long n)
{
    long long i = (long long)blockIdx.x * blockDim.x + threadIdx.x;
    if (i >= n) return;
    int c = (int)((i / (224 * 224)) % 3);
    float mean = (c == 0) ? 0.485f : (c == 1) ? 0.456f : 0.406f;
    float stdv = (c == 0) ? 0.229f : (c == 1) ? 0.224f : 0.225f;
    out[i] = pack_split(quant13((x[i] - mean) / stdv));
}

// 3x3 s2 pad1 maxpool on split pairs (max commutes with the monotone quantizer)
__global__ void maxpool_kernel(const uint32_t* __restrict__ in, uint32_t* __restrict__ out)
{
    long long i = (long long)blockIdx.x * blockDim.x + threadIdx.x;
    const long long n = 16LL * 64 * 56 * 56;
    if (i >= n) return;
    int ow = (int)(i % 56); long long t = i / 56;
    int oh = (int)(t % 56); t /= 56;
    int c  = (int)(t % 64); int b = (int)(t / 64);
    const uint32_t* p = in + (size_t)(b * 64 + c) * 112 * 112;
    float best = -3.4e38f; uint32_t bestp = 0u;
    for (int kh = 0; kh < 3; ++kh) {
        int ih = oh * 2 - 1 + kh;
        if ((unsigned)ih >= 112u) continue;
        for (int kw = 0; kw < 3; ++kw) {
            int iw = ow * 2 - 1 + kw;
            if ((unsigned)iw >= 112u) continue;
            uint32_t pv = p[ih * 112 + iw];
            float v = unpack_pair(pv);
            if (v > best) { best = v; bestp = pv; }
        }
    }
    out[i] = bestp;
}

// global avg pool over 7x7 (f32, unquantized input), then quant(16,13)+split
__global__ void avgpool_kernel(const float* __restrict__ in, uint32_t* __restrict__ out)
{
    int i = blockIdx.x * blockDim.x + threadIdx.x;   // b*512 + c
    if (i >= 16 * 512) return;
    const float* p = in + (size_t)i * 49;
    float s = 0.0f;
    for (int j = 0; j < 49; ++j) s += p[j];
    out[i] = pack_split(quant13(s * (1.0f / 49.0f)));
}

// ---------------------------------------------------------------------------
// Workspace layout (all slots 4 bytes/element)
// ---------------------------------------------------------------------------
static constexpr size_t NB = 16;
static constexpr size_t N_PIN  = NB * 3 * 224 * 224;     // input pairs
static constexpr size_t N_A0   = NB * 64 * 112 * 112;    // conv0 out pairs
static constexpr size_t N_A1   = NB * 64 * 56 * 56;      // maxpool out pairs
static constexpr size_t SLOT   = NB * 64 * 56 * 56;      // max per-stage tensor (stage1)
static constexpr size_t OFF_PIN  = 0;
static constexpr size_t OFF_A0   = OFF_PIN  + N_PIN;
static constexpr size_t OFF_A1   = OFF_A0   + N_A0;
static constexpr size_t OFF_R32  = OFF_A1   + N_A1;      // f32 residual (resconv)
static constexpr size_t OFF_RES2 = OFF_R32  + SLOT;      // f32 residual (block2)
static constexpr size_t OFF_P1   = OFF_RES2 + SLOT;
static constexpr size_t OFF_P2   = OFF_P1   + SLOT;
static constexpr size_t OFF_P3   = OFF_P2   + SLOT;
static constexpr size_t OFF_OUTA = OFF_P3   + SLOT;
static constexpr size_t OFF_OUTB = OFF_OUTA + SLOT;
static constexpr size_t OFF_O32  = OFF_OUTB + SLOT;      // f32 stage output (avgpool feed)
static constexpr size_t OFF_AVG  = OFF_O32  + SLOT;
static constexpr size_t OFF_W    = OFF_AVG  + NB * 512;  // repacked weights start here

extern "C" void kernel_launch(void* const* d_in, const int* in_sizes, int n_in,
                              void* d_out, int out_size, void* d_ws, size_t ws_size,
                              hipStream_t stream)
{
    (void)in_sizes; (void)n_in; (void)out_size; (void)ws_size;
    uint32_t* ws  = (uint32_t*)d_ws;
    float*    wsf = (float*)d_ws;

    const float* x    = (const float*)d_in[0];
    const float* w0   = (const float*)d_in[1];
    const float* bn0s = (const float*)d_in[2];
    const float* bn0b = (const float*)d_in[3];
    const float* fcw  = (const float*)d_in[32];
    const float* fcb  = (const float*)d_in[33];

    // ---- weight repack (sequential bump allocation in ws) ----
    size_t woff = OFF_W;
    auto repack = [&](const float* w, int Cout, int K, int CoutPad, int Kpad) -> uint32_t* {
        uint32_t* dst = ws + woff;
        long long n = (long long)CoutPad * Kpad;
        woff += (size_t)n;
        repack_kernel<<<dim3((unsigned)((n + 255) / 256)), dim3(256), 0, stream>>>(w, dst, Cout, K, Kpad, n);
        return dst;
    };

    const int cins[4] = { 64, 64, 128, 256 };
    const int cs[4]   = { 64, 128, 256, 512 };
    const int Hin[4]  = { 56, 56, 28, 14 };
    const int strd[4] = { 1, 2, 2, 2 };

    uint32_t* W0 = repack(w0, 64, 147, 64, 160);     // K=3*49 padded to 160
    uint32_t *WR[4], *WA[4], *WB[4], *WC[4], *WD[4];
    for (int i = 0; i < 4; ++i) {
        const int ci = cins[i], c = cs[i];
        WR[i] = repack((const float*)d_in[4 + 7 * i + 0], c, ci,     c, ci);
        WA[i] = repack((const float*)d_in[4 + 7 * i + 1], c, ci * 9, c, ci * 9);
        WB[i] = repack((const float*)d_in[4 + 7 * i + 2], c, c * 9,  c, c * 9);
        WC[i] = repack((const float*)d_in[4 + 7 * i + 3], c, c * 9,  c, c * 9);
        WD[i] = repack((const float*)d_in[4 + 7 * i + 4], c, c * 9,  c, c * 9);
    }
    uint32_t* WFC = repack(fcw, 1000, 512, 1024, 512);   // M padded to 64-multiple

    auto conv = [&](const uint32_t* actp, const uint32_t* wgt,
                    const float* bnS, const float* bnB, const float* res,
                    float* out32, uint32_t* outPair,
                    int Cin, int IH, int IW, int Cout, int CoutPad, int OH, int OW,
                    int KH, int KW, int s, int p, int Kdim, int Kpad, int flags) {
        int nTiles = (int)((NB * (size_t)OH * OW) / 16);
        dim3 grid((unsigned)((CoutPad / 64) * nTiles)), blk(32);
        if (KH == 1)
            conv_wmma_kernel<1, 1><<<grid, blk, 0, stream>>>(actp, wgt, bnS, bnB, res, out32, outPair,
                Cin, IH, IW, Cout, OH, OW, s, p, Kdim, Kpad, nTiles, flags);
        else if (KH == 3)
            conv_wmma_kernel<3, 3><<<grid, blk, 0, stream>>>(actp, wgt, bnS, bnB, res, out32, outPair,
                Cin, IH, IW, Cout, OH, OW, s, p, Kdim, Kpad, nTiles, flags);
        else
            conv_wmma_kernel<7, 7><<<grid, blk, 0, stream>>>(actp, wgt, bnS, bnB, res, out32, outPair,
                Cin, IH, IW, Cout, OH, OW, s, p, Kdim, Kpad, nTiles, flags);
    };

    // ---- stem ----
    preprocess_kernel<<<dim3((unsigned)((N_PIN + 255) / 256)), dim3(256), 0, stream>>>(x, ws + OFF_PIN, (long long)N_PIN);
    // conv0 7x7 s2 p3 + bn + relu -> pairs
    conv(ws + OFF_PIN, W0, bn0s, bn0b, nullptr, nullptr, ws + OFF_A0,
         3, 224, 224, 64, 64, 112, 112, 7, 7, 2, 3, 147, 160, F_BN | F_RELU | F_WPAIR);
    maxpool_kernel<<<dim3((unsigned)((N_A1 + 255) / 256)), dim3(256), 0, stream>>>(ws + OFF_A0, ws + OFF_A1);

    // ---- 4 stages of 2 basic blocks ----
    float*    R32  = wsf + OFF_R32;
    float*    RES2 = wsf + OFF_RES2;
    uint32_t* P1 = ws + OFF_P1, *P2 = ws + OFF_P2, *P3 = ws + OFF_P3;
    float*    O32 = wsf + OFF_O32;
    uint32_t* outSlots[2] = { ws + OFF_OUTA, ws + OFF_OUTB };
    const uint32_t* in = ws + OFF_A1;

    for (int i = 0; i < 4; ++i) {
        const int ci = cins[i], c = cs[i], H = Hin[i], s = strd[i], O = H / s;
        const float* bns = (const float*)d_in[4 + 7 * i + 5];
        const float* bnb = (const float*)d_in[4 + 7 * i + 6];
        uint32_t* OUT = outSlots[i & 1];
        // res = bn0(conv1x1(x))                      -> f32
        conv(in, WR[i], bns + 0 * c, bnb + 0 * c, nullptr, R32, nullptr,
             ci, H, H, c, c, O, O, 1, 1, s, 0, ci, ci, F_BN | F_W32);
        // o = relu(bn1(conv3x3(x)))                  -> pairs
        conv(in, WA[i], bns + 1 * c, bnb + 1 * c, nullptr, nullptr, P1,
             ci, H, H, c, c, O, O, 3, 3, s, 1, ci * 9, ci * 9, F_BN | F_RELU | F_WPAIR);
        // o = relu(bn2(conv3x3(o)) + res)            -> f32 + pairs
        conv(P1, WB[i], bns + 2 * c, bnb + 2 * c, R32, RES2, P2,
             c, O, O, c, c, O, O, 3, 3, 1, 1, c * 9, c * 9, F_BN | F_ADD | F_RELU | F_W32 | F_WPAIR);
        // o = relu(bn3(conv3x3(o)))                  -> pairs
        conv(P2, WC[i], bns + 3 * c, bnb + 3 * c, nullptr, nullptr, P3,
             c, O, O, c, c, O, O, 3, 3, 1, 1, c * 9, c * 9, F_BN | F_RELU | F_WPAIR);
        // out = relu(bn4(conv3x3(o)) + res2)         -> f32 + pairs
        conv(P3, WD[i], bns + 4 * c, bnb + 4 * c, RES2, O32, OUT,
             c, O, O, c, c, O, O, 3, 3, 1, 1, c * 9, c * 9, F_BN | F_ADD | F_RELU | F_W32 | F_WPAIR);
        in = OUT;
    }

    // ---- head ----
    avgpool_kernel<<<dim3(32), dim3(256), 0, stream>>>(O32, ws + OFF_AVG);
    // fc: quant(xq @ wq.T + b, 32, 24) -> d_out [16,1000]
    conv(ws + OFF_AVG, WFC, nullptr, fcb, nullptr, (float*)d_out, nullptr,
         512, 1, 1, 1000, 1024, 1, 1, 1, 1, 1, 0, 512, 512, F_FC);
}
